// BenchmarkLSTM_62732292325417
// MI455X (gfx1250) — matrix operand order repaired and checked by reading, not compiled, not run on hardware
//
#include <hip/hip_runtime.h>
#include <hip/hip_bf16.h>

// ---------------------------------------------------------------------------
// Persistent LSTM for MI455X (gfx1250, wave32).
//   - 32 workgroups, each owns a 16-row batch tile for the whole sequence.
//   - Concatenated [W_ih ; W_hh] pre-permuted into WMMA B-fragment order in
//     LDS once (padded to 32 column tiles so every wave owns exactly 4 tiles,
//     fully unrolled -> accumulators stay in fixed VGPRs).
//   - gates = [x_t | h] @ Wcat^T via v_wmma_f32_16x16x32_bf16 (f32 accum),
//     bias pre-folded into accumulator init.
//   - 2 workgroup barriers per timestep (x_{t+1} staged during cell update).
//   - f32 cell update with exp/rcp-based sigmoid/tanh; c lives in registers.
// ---------------------------------------------------------------------------

typedef __attribute__((ext_vector_type(16))) __bf16 v16bf;
typedef __attribute__((ext_vector_type(8)))  float  v8f;

union BF16Frag { uint4 u4[2]; v16bf v; };

__device__ __forceinline__ unsigned short f2bf(float f) {
    unsigned u = __float_as_uint(f);
    unsigned r = (u + 0x7FFFu + ((u >> 16) & 1u)) >> 16;  // RNE truncation
    return (unsigned short)r;
}

__device__ __forceinline__ float fast_sigmoid(float x) {
    return __builtin_amdgcn_rcpf(1.0f + __expf(-x));
}
__device__ __forceinline__ float fast_tanh(float x) {
    // 1 - 2/(e^{2x}+1): saturates correctly at +/-1
    return 1.0f - 2.0f * __builtin_amdgcn_rcpf(__expf(2.0f * x) + 1.0f);
}

#define HID     100
#define GATESN  400        // 4*HID real columns
#define NTPAD   32         // column tiles padded 25 -> 32 (4 per wave, static)
#define GSTRIDE 404        // gate LDS row stride (f32) - bank-conflict pad
#define HPAD    128        // hidden K padded to chunk multiple

// X: [S, B, KIN] f32 layer input. Hout: [S, B, 100] f32.
template <int KIN, int KINPAD>
__global__ __launch_bounds__(256, 1)
void lstm_layer_kernel(const float* __restrict__ X,
                       const float* __restrict__ Wih,
                       const float* __restrict__ Whh,
                       const float* __restrict__ bih,
                       const float* __restrict__ bhh,
                       float* __restrict__ Hout,
                       int S, int Bn) {
    constexpr int KPAD    = KINPAD + HPAD;   // 192 (layer0) / 256 (layers 1,2)
    constexpr int NCH     = KPAD / 32;       // K chunks of 32
    constexpr int ASTRIDE = KPAD + 8;        // A row stride in bf16 elems

    extern __shared__ __align__(16) char smem[];
    // [0] B fragments, pre-permuted: word index ((kt*32+nt)*32+lane)*8 + j
    unsigned*       Bw    = (unsigned*)smem;
    constexpr size_t bfW  = (size_t)NCH * NTPAD * 32 * 8;            // words
    // [1] A buffer: bf16 [16 rows][ASTRIDE] = x_t (0..KINPAD) | h (KINPAD..)
    unsigned short* sA    = (unsigned short*)(smem + bfW * 4);
    // [2] gate accumulators f32 [16][GSTRIDE]
    float*          gatesL = (float*)(smem + bfW * 4 + (size_t)16 * ASTRIDE * 2);
    // [3] bias f32 [400]
    float*          biasL  = gatesL + 16 * GSTRIDE;

    const int tid  = threadIdx.x;
    const int lane = tid & 31;
    const int wv   = tid >> 5;
    const int b0   = blockIdx.x * 16;

    // ---- one-time init -----------------------------------------------------
    for (int e = tid; e < GATESN; e += 256) biasL[e] = bih[e] + bhh[e];
    for (int e = tid; e < 16 * ASTRIDE; e += 256) sA[e] = 0;  // h0=0, pads=0
    // Pre-permute weights into B-fragment order (CDNA5 16-bit B 32x16):
    //   lane<16:  K = kt*32 + {0-7,16-23} (word j -> pair);  lane>=16: K = kt*32 + {8-15,24-31};
    //   column n = nt*16 + (lane&15).  Pad tiles (n>=400) and pad K are zero.
    {
        constexpr int total = NCH * NTPAD * 32 * 8;
        for (int e = tid; e < total; e += 256) {
            int j  = e & 7;
            int ln = (e >> 3) & 31;
            int nt = (e >> 8) & 31;
            int kt = (e >> 8) >> 5;
            int n  = nt * 16 + (ln & 15);
            int k  = kt * 32 + ((ln < 16) ? 0 : 8) + 2 * j + ((j < 4) ? 0 : 8);
            unsigned short h0 = 0, h1 = 0;
            if (n < GATESN) {
                #pragma unroll
                for (int half = 0; half < 2; ++half) {
                    int kk = k + half;
                    float w;
                    if (kk < KINPAD) w = (kk < KIN) ? Wih[(size_t)n * KIN + kk] : 0.0f;
                    else {
                        int kh = kk - KINPAD;
                        w = (kh < HID) ? Whh[(size_t)n * HID + kh] : 0.0f;
                    }
                    if (half == 0) h0 = f2bf(w); else h1 = f2bf(w);
                }
            }
            Bw[e] = (unsigned)h0 | ((unsigned)h1 << 16);
        }
    }
    __syncthreads();

    // per-wave tile bias values (time-invariant) + stage x_0
    const int cbase = (lane & 15);
    const int rbase = (lane < 16) ? 0 : 8;
    float tb[4];
    #pragma unroll
    for (int i = 0; i < 4; ++i) {
        const int nt = wv + 8 * i;
        tb[i] = (nt < 25) ? biasL[nt * 16 + cbase] : 0.0f;
    }
    {
        const float* xb = X + (size_t)b0 * KIN;
        #pragma unroll 1
        for (int e = tid; e < 16 * KIN; e += 256) {
            int m = e / KIN, d = e - m * KIN;
            sA[m * ASTRIDE + d] = f2bf(xb[m * KIN + d]);
        }
    }
    // per-thread slice of the cell state (16*100 elems over 256 threads)
    float creg[7];
    #pragma unroll
    for (int i = 0; i < 7; ++i) creg[i] = 0.0f;

    const uint4* Bq = (const uint4*)Bw;
    __syncthreads();

    // ---- time loop (2 barriers per step) -----------------------------------
    for (int t = 0; t < S; ++t) {
        // phase 1: gates = [x_t | h] @ Wcat^T (bias folded into acc init)
        v8f acc0, acc1, acc2, acc3;
        #pragma unroll
        for (int r = 0; r < 8; ++r) {
            acc0[r] = tb[0]; acc1[r] = tb[1]; acc2[r] = tb[2]; acc3[r] = tb[3];
        }
        #pragma unroll 1
        for (int kt = 0; kt < NCH; ++kt) {
            // A fragment (16x32): lanes 0-15 row=lane, K 0-7/16-23;
            //                     lanes 16-31 row=lane-16, K 8-15/24-31
            BF16Frag a;
            const int ka = (lane & 15) * ASTRIDE + kt * 32;
            a.u4[0] = *(const uint4*)(sA + ka + ((lane < 16) ? 0 : 8));
            a.u4[1] = *(const uint4*)(sA + ka + ((lane < 16) ? 16 : 24));
            BF16Frag b0f, b1f, b2f, b3f;
            const size_t qb = ((size_t)(kt * NTPAD + wv) * 32 + lane) * 2;
            b0f.u4[0] = Bq[qb + 0 * 512];      b0f.u4[1] = Bq[qb + 0 * 512 + 1];
            b1f.u4[0] = Bq[qb + 1 * 512];      b1f.u4[1] = Bq[qb + 1 * 512 + 1];
            b2f.u4[0] = Bq[qb + 2 * 512];      b2f.u4[1] = Bq[qb + 2 * 512 + 1];
            b3f.u4[0] = Bq[qb + 3 * 512];      b3f.u4[1] = Bq[qb + 3 * 512 + 1];
            acc0 = __builtin_amdgcn_wmma_f32_16x16x32_bf16(false, a.v, false, b0f.v, (short)0, acc0, false, false);
            acc1 = __builtin_amdgcn_wmma_f32_16x16x32_bf16(false, a.v, false, b1f.v, (short)0, acc1, false, false);
            acc2 = __builtin_amdgcn_wmma_f32_16x16x32_bf16(false, a.v, false, b2f.v, (short)0, acc2, false, false);
            acc3 = __builtin_amdgcn_wmma_f32_16x16x32_bf16(false, a.v, false, b3f.v, (short)0, acc3, false, false);
        }
        // spill real gate tiles.  C/D layout: VGPR r -> row r (lanes 0-15)
        // or 8+r (lanes 16-31); N = nt*16 + (lane&15).
        #pragma unroll
        for (int i = 0; i < 4; ++i) {
            const int nt = wv + 8 * i;
            if (nt < 25) {                          // wave-uniform; pads skipped
                const int col = nt * 16 + cbase;
                const v8f av = (i == 0) ? acc0 : (i == 1) ? acc1 : (i == 2) ? acc2 : acc3;
                #pragma unroll
                for (int r = 0; r < 8; ++r)
                    gatesL[(rbase + r) * GSTRIDE + col] = av[r];
            }
        }
        __syncthreads();

        // phase 2: cell update + stage x_{t+1}
        {
            float* hb = Hout + ((size_t)t * Bn + b0) * HID;
            #pragma unroll
            for (int ci = 0; ci < 7; ++ci) {
                int e = tid + ci * 256;
                if (e < 16 * HID) {
                    int m = e / HID, j = e - m * HID;
                    const float* gr = gatesL + m * GSTRIDE;
                    float ig = fast_sigmoid(gr[j]);
                    float fg = fast_sigmoid(gr[HID + j]);
                    float gg = fast_tanh(gr[2 * HID + j]);
                    float og = fast_sigmoid(gr[3 * HID + j]);
                    float cn = fg * creg[ci] + ig * gg;
                    creg[ci] = cn;
                    float hn = og * fast_tanh(cn);
                    sA[m * ASTRIDE + KINPAD + j] = f2bf(hn);
                    hb[m * HID + j] = hn;
                }
            }
            if (t + 1 < S) {
                const float* xb = X + ((size_t)(t + 1) * Bn + b0) * KIN;
                #pragma unroll 1
                for (int e = tid; e < 16 * KIN; e += 256) {
                    int m = e / KIN, d = e - m * KIN;
                    sA[m * ASTRIDE + d] = f2bf(xb[m * KIN + d]);
                }
            }
        }
        __syncthreads();
    }
}

// out[row, o] = h[row,:] . W_lin[o,:] + b_lin[o]   (N=8, trivially small)
__global__ __launch_bounds__(256)
void linear_kernel(const float* __restrict__ H, const float* __restrict__ Wl,
                   const float* __restrict__ bl, float* __restrict__ out,
                   long long rows) {
    long long idx = (long long)blockIdx.x * 256 + threadIdx.x;
    if (idx >= rows * 8) return;
    int o = (int)(idx & 7);
    long long row = idx >> 3;
    float a = bl[o];
    const float* hr = H + row * HID;
    const float* wr = Wl + o * HID;
    #pragma unroll 4
    for (int k = 0; k < HID; ++k) a += hr[k] * wr[k];
    out[row * 8 + o] = a;
}

static constexpr size_t lds_bytes(int KPAD) {
    return (size_t)(KPAD / 32) * NTPAD * 32 * 8 * 4   // B fragments
         + (size_t)16 * (KPAD + 8) * 2                // A buffer
         + (size_t)16 * GSTRIDE * 4                   // gates
         + (size_t)GATESN * 4;                        // bias
}

extern "C" void kernel_launch(void* const* d_in, const int* in_sizes, int n_in,
                              void* d_out, int out_size, void* d_ws, size_t ws_size,
                              hipStream_t stream) {
    (void)in_sizes; (void)n_in; (void)out_size; (void)ws_size;
    const int S = 672, B = 512;
    const float* x     = (const float*)d_in[0];
    const float* Wih0  = (const float*)d_in[1];
    const float* Whh0  = (const float*)d_in[2];
    const float* bih0  = (const float*)d_in[3];
    const float* bhh0  = (const float*)d_in[4];
    const float* Wih1  = (const float*)d_in[5];
    const float* Whh1  = (const float*)d_in[6];
    const float* bih1  = (const float*)d_in[7];
    const float* bhh1  = (const float*)d_in[8];
    const float* Wih2  = (const float*)d_in[9];
    const float* Whh2  = (const float*)d_in[10];
    const float* bih2  = (const float*)d_in[11];
    const float* bhh2  = (const float*)d_in[12];
    const float* Wlin  = (const float*)d_in[13];
    const float* blin  = (const float*)d_in[14];

    float* Ha = (float*)d_ws;                          // [S,B,100]
    float* Hb = Ha + (size_t)S * B * HID;              // [S,B,100]

    const size_t sh0 = lds_bytes(64 + HPAD);    // layer 0: Kin=38 -> KinPad=64
    const size_t sh1 = lds_bytes(HPAD + HPAD);  // layers 1/2: KinPad=128

    (void)hipFuncSetAttribute((const void*)&lstm_layer_kernel<38, 64>,
                              hipFuncAttributeMaxDynamicSharedMemorySize, (int)sh0);
    (void)hipFuncSetAttribute((const void*)&lstm_layer_kernel<100, 128>,
                              hipFuncAttributeMaxDynamicSharedMemorySize, (int)sh1);

    dim3 grid(B / 16), block(256);
    lstm_layer_kernel<38, 64><<<grid, block, sh0, stream>>>(x,  Wih0, Whh0, bih0, bhh0, Ha, S, B);
    lstm_layer_kernel<100, 128><<<grid, block, sh1, stream>>>(Ha, Wih1, Whh1, bih1, bhh1, Hb, S, B);
    lstm_layer_kernel<100, 128><<<grid, block, sh1, stream>>>(Hb, Wih2, Whh2, bih2, bhh2, Ha, S, B);

    long long rows = (long long)S * B;
    long long tot  = rows * 8;
    linear_kernel<<<dim3((unsigned)((tot + 255) / 256)), block, 0, stream>>>(Ha, Wlin, blin, (float*)d_out, rows);
}
